// SGSLanguageModel_10814727651383
// MI455X (gfx1250) — matrix operand order repaired and checked by reading, not compiled, not run on hardware
//
#include <hip/hip_runtime.h>
#include <hip/hip_bf16.h>

#define B_  4
#define L_  512
#define BL_ 2048
#define DS_ 128
#define HH_ 4
#define DF_ 1000
#define D2_ 2000
#define D4_ 4000
#define VV_ 32000

typedef __attribute__((ext_vector_type(16))) _Float16 v16h;
typedef __attribute__((ext_vector_type(8)))  float    v8f;
typedef __attribute__((ext_vector_type(4)))  unsigned int u32x4;
typedef __attribute__((ext_vector_type(8)))  unsigned int u32x8;

// ---------------------------------------------------------------------------
// WMMA fragment loaders (f32 memory -> f16 fragments), TN layout.
// Fast versions: caller guarantees k0+32 <= K (no guards, pure b128 loads).
// ---------------------------------------------------------------------------
__device__ __forceinline__ v16h load_a_fast(const float* __restrict__ Arow,
                                            int k0) {
  int lane = threadIdx.x & 31;
  int kb = k0 + ((lane & 16) ? 8 : 0);
  const float4* p0 = (const float4*)(Arow + kb);
  const float4* p1 = (const float4*)(Arow + kb + 16);
  float4 x0 = p0[0], x1 = p0[1], y0 = p1[0], y1 = p1[1];
  v16h a;
  a[0]=(_Float16)x0.x; a[1]=(_Float16)x0.y; a[2]=(_Float16)x0.z; a[3]=(_Float16)x0.w;
  a[4]=(_Float16)x1.x; a[5]=(_Float16)x1.y; a[6]=(_Float16)x1.z; a[7]=(_Float16)x1.w;
  a[8]=(_Float16)y0.x; a[9]=(_Float16)y0.y; a[10]=(_Float16)y0.z; a[11]=(_Float16)y0.w;
  a[12]=(_Float16)y1.x; a[13]=(_Float16)y1.y; a[14]=(_Float16)y1.z; a[15]=(_Float16)y1.w;
  return a;
}

__device__ __forceinline__ v16h load_b_fast(const float* __restrict__ Brow,
                                            int k0) {
  int lane = threadIdx.x & 31;
  int kb = k0 + ((lane & 16) ? 8 : 0);
  const float4* p = (const float4*)(Brow + kb);
  const float4* q = (const float4*)(Brow + kb + 16);
  float4 x0 = p[0], x1 = p[1], x2 = q[0], x3 = q[1];
  v16h b;
  b[0]=(_Float16)x0.x; b[1]=(_Float16)x0.y; b[2]=(_Float16)x0.z; b[3]=(_Float16)x0.w;
  b[4]=(_Float16)x1.x; b[5]=(_Float16)x1.y; b[6]=(_Float16)x1.z; b[7]=(_Float16)x1.w;
  b[8]=(_Float16)x2.x; b[9]=(_Float16)x2.y; b[10]=(_Float16)x2.z; b[11]=(_Float16)x2.w;
  b[12]=(_Float16)x3.x; b[13]=(_Float16)x3.y; b[14]=(_Float16)x3.z; b[15]=(_Float16)x3.w;
  return b;
}

__device__ __forceinline__ v16h load_a_guard(const float* __restrict__ Arow,
                                             int k0, int K) {
  int lane = threadIdx.x & 31;
  int kb = k0 + ((lane & 16) ? 8 : 0);
  v16h a;
#pragma unroll
  for (int i = 0; i < 8; ++i) {
    int k = kb + i;
    a[i] = (k < K) ? (_Float16)Arow[k] : (_Float16)0.f;
  }
#pragma unroll
  for (int i = 0; i < 8; ++i) {
    int k = kb + 16 + i;
    a[8 + i] = (k < K) ? (_Float16)Arow[k] : (_Float16)0.f;
  }
  return a;
}

__device__ __forceinline__ v16h load_b_guard(const float* __restrict__ Brow,
                                             int k0, int K) {
  int lane = threadIdx.x & 31;
  int kb = k0 + ((lane & 16) ? 8 : 0);
  v16h b;
#pragma unroll
  for (int i = 0; i < 16; ++i) {
    int k = kb + i + (i >= 8 ? 8 : 0);
    b[i] = (k < K) ? (_Float16)Brow[k] : (_Float16)0.f;
  }
  return b;
}

// ---------------------------------------------------------------------------
// Generic TN WMMA GEMM: C[m,n] = epi( sum_k A[m,k]*W[n,k] + bias[n] )
// Wave tile: 32(M) x 64(N) = 8 WMMA accumulators; block = 8 waves = 256(M).
// Batched over blockIdx.z: A,C use z; W,bias,base use z/bdiv.
// epi: 0 none; 1 base+tanh; 2 exact gelu; 3 base+v; 4 attention eff.
// ---------------------------------------------------------------------------
__global__ void __launch_bounds__(256) wmma_gemm_tn(
    const float* __restrict__ A, long sA,
    const float* __restrict__ Bw, long sB, int bdiv,
    const float* __restrict__ bias, long sBias,
    const float* __restrict__ base, long sBase, int ldBase,
    float* __restrict__ C, long sCq, long sCr,
    int ldA, int ldB, int ldC,
    int M, int N, int K, int epi, const float* __restrict__ p_logtau) {
  int z = blockIdx.z;
  int zq = z / bdiv, zr = z % bdiv;
  A += (long)z * sA;
  Bw += (long)zq * sB;
  if (bias) bias += (long)zq * sBias;
  if (base) base += (long)zq * sBase;
  C += (long)zq * sCq + (long)zr * sCr;

  int wave = threadIdx.x >> 5, lane = threadIdx.x & 31;
  int lane15 = lane & 15;
  int m0 = (blockIdx.y * 8 + wave) * 32;
  int n0 = blockIdx.x * 64;

  const float* Arow0 = A + (long)(m0 + lane15) * ldA;
  const float* Arow1 = A + (long)(m0 + 16 + lane15) * ldA;
  const float* Brow[4];
#pragma unroll
  for (int t = 0; t < 4; ++t) {
    int nn = n0 + t * 16 + lane15;
    if (nn > N - 1) nn = N - 1;
    Brow[t] = Bw + (long)nn * ldB;
  }
  v8f acc[2][4] = {};
  int k0 = 0;
  for (; k0 + 32 <= K; k0 += 32) {
    v16h af0 = load_a_fast(Arow0, k0);
    v16h af1 = load_a_fast(Arow1, k0);
#pragma unroll
    for (int t = 0; t < 4; ++t) {
      v16h bf = load_b_fast(Brow[t], k0);
      acc[0][t] = __builtin_amdgcn_wmma_f32_16x16x32_f16(
          false, af0, false, bf, (short)0, acc[0][t], false, false);
      acc[1][t] = __builtin_amdgcn_wmma_f32_16x16x32_f16(
          false, af1, false, bf, (short)0, acc[1][t], false, false);
    }
  }
  if (k0 < K) {  // single guarded tail step (K % 32 != 0)
    v16h af0 = load_a_guard(Arow0, k0, K);
    v16h af1 = load_a_guard(Arow1, k0, K);
#pragma unroll
    for (int t = 0; t < 4; ++t) {
      v16h bf = load_b_guard(Brow[t], k0, K);
      acc[0][t] = __builtin_amdgcn_wmma_f32_16x16x32_f16(
          false, af0, false, bf, (short)0, acc[0][t], false, false);
      acc[1][t] = __builtin_amdgcn_wmma_f32_16x16x32_f16(
          false, af1, false, bf, (short)0, acc[1][t], false, false);
    }
  }

  float escale = (epi == 4) ? 0.5f * __expf(-p_logtau[0]) : 0.f;
#pragma unroll
  for (int s = 0; s < 2; ++s) {
    int rbase = m0 + s * 16 + ((lane >> 4) << 3);
#pragma unroll
    for (int t = 0; t < 4; ++t) {
      int n = n0 + t * 16 + lane15;
      if (n >= N) continue;
      float bv = bias ? bias[n] : 0.f;
#pragma unroll
      for (int r = 0; r < 8; ++r) {
        int m = rbase + r;
        float v = acc[s][t][r];
        float o;
        if (epi == 0) {
          o = v + bv;
        } else if (epi == 1) {
          o = base[(long)m * ldBase + n] + tanhf(v + bv);
        } else if (epi == 2) {
          float xx = v + bv;
          o = 0.5f * xx * (1.f + erff(xx * 0.70710678f));
        } else if (epi == 3) {
          o = base[(long)m * ldBase + n] + v + bv;
        } else {  // 4: eff = clip(alpha[j]*exp(-0.5*(v+mu2[j])/tau)*mask)
          float mah = v + base[n];
          float ef = bv * __expf(-escale * mah);
          ef = (n <= m) ? fminf(fmaxf(ef, 0.f), 0.999999f) : 0.f;
          o = ef;
        }
        C[(long)m * ldC + n] = o;
      }
    }
  }
}

// ---------------------------------------------------------------------------
// Embedding gather: mu = tok_mu[tok]+pos_mu[l]; iv = exp(-log_var);
// alpha0 = sigmoid(raw); feats gather.
// ---------------------------------------------------------------------------
__global__ void __launch_bounds__(256) embed_kernel(
    const int* __restrict__ tokens,
    const float* __restrict__ tok_mu, const float* __restrict__ tok_log_var,
    const float* __restrict__ tok_raw_alpha, const float* __restrict__ tok_features,
    const float* __restrict__ pos_mu,
    float* __restrict__ mu, float* __restrict__ ivb,
    float* __restrict__ alpha, float* __restrict__ feats) {
  long row = blockIdx.x;
  long l = row % L_;
  long tok = tokens[row];
  for (int i = threadIdx.x; i < DS_; i += 256) {
    mu[row * DS_ + i] = tok_mu[tok * DS_ + i] + pos_mu[l * DS_ + i];
    ivb[row * DS_ + i] = __expf(-tok_log_var[tok * DS_ + i]);
  }
  if (threadIdx.x == 0)
    alpha[row] = 1.f / (1.f + __expf(-tok_raw_alpha[tok]));
  for (int i = threadIdx.x; i < DF_; i += 256)
    feats[row * DF_ + i] = tok_features[tok * DF_ + i];
}

// Bhat[b*L+j, 0:128]=iv, [128:256]=-2*mu*iv; mu2[b*L+j]=sum mu^2*iv
__global__ void __launch_bounds__(128) prep_attn(
    const float* __restrict__ mu, const float* __restrict__ ivb,
    float* __restrict__ Bhat, float* __restrict__ mu2) {
  __shared__ float sm[4];
  long row = blockIdx.x;
  int d = threadIdx.x;
  float m = mu[row * DS_ + d], iv = ivb[row * DS_ + d];
  Bhat[row * 256 + d] = iv;
  Bhat[row * 256 + 128 + d] = -2.f * m * iv;
  float p = m * m * iv;
#pragma unroll
  for (int o = 16; o > 0; o >>= 1) p += __shfl_down(p, o);
  if ((d & 31) == 0) sm[d >> 5] = p;
  __syncthreads();
  if (d == 0) mu2[row] = sm[0] + sm[1] + sm[2] + sm[3];
}

// Ahat[(b*H+h)*L+l, 0:128]=q^2, [128:256]=q  (q from qbuf[row,512])
__global__ void __launch_bounds__(256) ahat_build(
    const float* __restrict__ qbuf, float* __restrict__ Ahat) {
  long row = blockIdx.x;  // b*L + l
  long b = row / L_, l = row % L_;
  for (int e = threadIdx.x; e < HH_ * DS_; e += 256) {
    int h = e >> 7, d = e & 127;
    float qv = qbuf[row * (HH_ * DS_) + e];
    long ar = ((b * HH_ + h) * L_ + l) * 256;
    Ahat[ar + d] = qv * qv;
    Ahat[ar + 128 + d] = qv;
  }
}

// Causal alpha-compositing scan: w[l,j] = eff * exp(cumsum_excl log1p(-eff)).
// One wave per row of 512, 16 contiguous elements per lane.
__global__ void __launch_bounds__(256) composite_scan(float* __restrict__ w) {
  long row = (long)blockIdx.x * 8 + (threadIdx.x >> 5);
  int lane = threadIdx.x & 31;
  float* p = w + row * (long)L_ + lane * 16;
  float e[16];
  float4* p4 = (float4*)p;
#pragma unroll
  for (int i = 0; i < 4; ++i) {
    float4 v = p4[i];
    e[i * 4 + 0] = v.x; e[i * 4 + 1] = v.y; e[i * 4 + 2] = v.z; e[i * 4 + 3] = v.w;
  }
  float lp[16], s = 0.f;
#pragma unroll
  for (int i = 0; i < 16; ++i) { lp[i] = s; s += log1pf(-e[i]); }
  float v = s;
#pragma unroll
  for (int off = 1; off < 32; off <<= 1) {
    float t = __shfl_up(v, off);
    if (lane >= off) v += t;
  }
  float excl = v - s;  // exclusive prefix of lane sums
#pragma unroll
  for (int i = 0; i < 16; ++i) e[i] = e[i] * __expf(excl + lp[i]);
#pragma unroll
  for (int i = 0; i < 4; ++i) {
    float4 o; o.x = e[i*4+0]; o.y = e[i*4+1]; o.z = e[i*4+2]; o.w = e[i*4+3];
    p4[i] = o;
  }
}

// ---------------------------------------------------------------------------
// feats [B,L,DF] -> featsT [B,DF,L] transpose, 64x64 tiles staged into LDS by
// the Tensor Data Mover (tensor_load_to_lds, 2D descriptor, TENSORcnt wait).
// OOB columns (f >= 1000) return zero per D# semantics.
// ---------------------------------------------------------------------------
#define TT 64
__global__ void __launch_bounds__(256) feats_transpose_tdm(
    const float* __restrict__ feats, float* __restrict__ featsT) {
  __shared__ float tile[TT * TT];
  int f0 = blockIdx.x * TT;
  int j0 = blockIdx.y * TT;
  int b = blockIdx.z;
  if (threadIdx.x < 32) {
    unsigned long long ga =
        (unsigned long long)(feats + ((long)b * L_ + j0) * DF_ + f0);
    u32x4 g0;
    u32x8 g1;
    g0[0] = 1u;                                  // count=1, user descriptor
    g0[1] = 0u;                                  // lds_addr (tile @ LDS 0)
    g0[2] = (unsigned)ga;                        // global_addr lo
    g0[3] = ((unsigned)(ga >> 32) & 0x01FFFFFFu) // global_addr hi (bits 56:32)
            | 0x80000000u;                       // type=2 (image)
    g1[0] = 2u << 16;                            // data_size=4B
    g1[1] = ((unsigned)DF_ & 0xFFFFu) << 16;     // tensor_dim0 lo16
    g1[2] = (((unsigned)DF_ >> 16) & 0xFFFFu) | (((unsigned)L_ & 0xFFFFu) << 16);
    g1[3] = (((unsigned)L_ >> 16) & 0xFFFFu) | ((unsigned)TT << 16);  // tile_dim0
    g1[4] = (unsigned)TT;                        // tile_dim1 (tile_dim2=0)
    g1[5] = (unsigned)DF_;                       // tensor_dim0_stride lo32
    g1[6] = 0u;
    g1[7] = 0u;
    asm volatile("tensor_load_to_lds %0, %1" ::"s"(g0), "s"(g1) : "memory");
    __builtin_amdgcn_s_wait_tensorcnt(0);
  }
  __syncthreads();
  for (int i = threadIdx.x; i < TT * TT; i += 256) {
    int jj = i / TT, ff = i % TT;
    int f = f0 + ff;
    if (f < DF_)
      featsT[((long)b * DF_ + f) * L_ + (j0 + jj)] = tile[jj * TT + ff];
  }
}

// x = concat(feats, meaning)
__global__ void __launch_bounds__(256) concat_x(
    const float* __restrict__ feats, const float* __restrict__ meaning,
    float* __restrict__ x) {
  long row = blockIdx.x;
  for (int i = threadIdx.x; i < DF_; i += 256) {
    x[row * D2_ + i] = feats[row * DF_ + i];
    x[row * D2_ + DF_ + i] = meaning[row * DF_ + i];
  }
}

// LayerNorm rows of width Wd
__global__ void __launch_bounds__(256) ln_rows(
    const float* __restrict__ x, const float* __restrict__ gw,
    const float* __restrict__ bw, float* __restrict__ out, int Wd) {
  __shared__ float sm[8];
  __shared__ float stat[2];
  long row = blockIdx.x;
  const float* xr = x + row * (long)Wd;
  float s = 0.f;
  for (int i = threadIdx.x; i < Wd; i += 256) s += xr[i];
#pragma unroll
  for (int o = 16; o > 0; o >>= 1) s += __shfl_down(s, o);
  if ((threadIdx.x & 31) == 0) sm[threadIdx.x >> 5] = s;
  __syncthreads();
  if (threadIdx.x == 0) {
    float t = 0.f;
    for (int i = 0; i < 8; ++i) t += sm[i];
    stat[0] = t / Wd;
  }
  __syncthreads();
  float mean = stat[0];
  float v = 0.f;
  for (int i = threadIdx.x; i < Wd; i += 256) {
    float d = xr[i] - mean;
    v += d * d;
  }
#pragma unroll
  for (int o = 16; o > 0; o >>= 1) v += __shfl_down(v, o);
  __syncthreads();
  if ((threadIdx.x & 31) == 0) sm[threadIdx.x >> 5] = v;
  __syncthreads();
  if (threadIdx.x == 0) {
    float t = 0.f;
    for (int i = 0; i < 8; ++i) t += sm[i];
    stat[1] = rsqrtf(t / Wd + 1e-5f);
  }
  __syncthreads();
  float rstd = stat[1];
  float* orow = out + row * (long)Wd;
  for (int i = threadIdx.x; i < Wd; i += 256)
    orow[i] = (xr[i] - mean) * rstd * gw[i] + bw[i];
}

// alpha[row] *= sigmoid(dot(x_row, Wg) + bg)
__global__ void __launch_bounds__(256) gate_alpha(
    const float* __restrict__ x, const float* __restrict__ Wg,
    const float* __restrict__ bg, float* __restrict__ alpha) {
  __shared__ float sm[8];
  long row = blockIdx.x;
  const float* xr = x + row * (long)D2_;
  float s = 0.f;
  for (int i = threadIdx.x; i < D2_; i += 256) s += xr[i] * Wg[i];
#pragma unroll
  for (int o = 16; o > 0; o >>= 1) s += __shfl_down(s, o);
  if ((threadIdx.x & 31) == 0) sm[threadIdx.x >> 5] = s;
  __syncthreads();
  if (threadIdx.x == 0) {
    float t = 0.f;
    for (int i = 0; i < 8; ++i) t += sm[i];
    alpha[row] *= 1.f / (1.f + __expf(-(t + bg[0])));
  }
}

// ---------------------------------------------------------------------------
extern "C" void kernel_launch(void* const* d_in, const int* in_sizes, int n_in,
                              void* d_out, int out_size, void* d_ws,
                              size_t ws_size, hipStream_t stream) {
  const int* tokens = (const int*)d_in[0];
  const float* tok_mu = (const float*)d_in[1];
  const float* tok_log_var = (const float*)d_in[2];
  const float* tok_raw_alpha = (const float*)d_in[3];
  const float* tok_features = (const float*)d_in[4];
  const float* pos_mu = (const float*)d_in[5];
  const float* log_tau = (const float*)d_in[6];
  const float* Wq = (const float*)d_in[7];
  const float* bq = (const float*)d_in[8];
  const float* Whead = (const float*)d_in[9];
  const float* bhead = (const float*)d_in[10];
  const float* Wmu = (const float*)d_in[11];
  const float* bmu = (const float*)d_in[12];
  const float* Wg = (const float*)d_in[13];
  const float* bg = (const float*)d_in[14];
  const float* ln_g = (const float*)d_in[15];
  const float* ln_b = (const float*)d_in[16];
  const float* W1 = (const float*)d_in[17];
  const float* b1 = (const float*)d_in[18];
  const float* W2 = (const float*)d_in[19];
  const float* b2 = (const float*)d_in[20];
  const float* lnf_g = (const float*)d_in[21];
  const float* lnf_b = (const float*)d_in[22];
  const float* Wlm = (const float*)d_in[23];
  float* out = (float*)d_out;

  float* ws = (float*)d_ws;
  float* mu = ws;                         // BL*128
  float* ivb = mu + (long)BL_ * DS_;      // BL*128
  float* alpha = ivb + (long)BL_ * DS_;   // BL
  float* mu2 = alpha + BL_;               // BL
  float* feats = mu2 + BL_;               // BL*1000
  float* qbuf = feats + (long)BL_ * DF_;  // BL*512
  float* Ahat = qbuf + (long)BL_ * 512;   // B*H*L*256
  float* Bhat = Ahat + (long)B_ * HH_ * L_ * 256;  // B*L*256
  float* wbuf = Bhat + (long)B_ * L_ * 256;        // B*H*L*L
  float* featsT = wbuf + (long)B_ * HH_ * L_ * L_; // B*1000*512
  float* mh = featsT + (long)B_ * DF_ * L_;        // BL*4000 (reused as ffn1)
  float* meaning = mh + (long)BL_ * D4_;           // BL*1000
  float* xbuf = meaning + (long)BL_ * DF_;         // BL*2000
  float* hbuf = xbuf + (long)BL_ * D2_;            // BL*2000
  float* yln = hbuf + (long)BL_ * D2_;             // BL*1000

  dim3 blk(256);

  embed_kernel<<<BL_, blk, 0, stream>>>(tokens, tok_mu, tok_log_var,
                                        tok_raw_alpha, tok_features, pos_mu,
                                        mu, ivb, alpha, feats);

  for (int p = 0; p < 3; ++p) {
    // q = mu @ Wq^T + bq   [BL,512]
    wmma_gemm_tn<<<dim3(8, 8, 1), blk, 0, stream>>>(
        mu, 0, Wq, 0, 1, bq, 0, nullptr, 0, 0, qbuf, 0, 0,
        DS_, DS_, HH_ * DS_, BL_, HH_ * DS_, DS_, 0, log_tau);
    ahat_build<<<BL_, blk, 0, stream>>>(qbuf, Ahat);
    prep_attn<<<BL_, dim3(128), 0, stream>>>(mu, ivb, Bhat, mu2);

    // eff = epi4( Ahat @ Bhat^T )  per (b,h):  [512,512]
    wmma_gemm_tn<<<dim3(8, 2, B_ * HH_), blk, 0, stream>>>(
        Ahat, (long)L_ * 256, Bhat, (long)L_ * 256, HH_,
        alpha, L_, mu2, L_, 0,
        wbuf, (long)HH_ * L_ * L_, (long)L_ * L_,
        256, 256, L_, L_, L_, 256, 4, log_tau);

    composite_scan<<<B_ * HH_ * L_ / 8, blk, 0, stream>>>(wbuf);

    feats_transpose_tdm<<<dim3(16, L_ / TT, B_), blk, 0, stream>>>(feats, featsT);

    // mh[b,l,h*1000+f] = w @ featsT^T  per (b,h): [512,1000]
    wmma_gemm_tn<<<dim3(16, 2, B_ * HH_), blk, 0, stream>>>(
        wbuf, (long)L_ * L_, featsT, (long)DF_ * L_, HH_,
        nullptr, 0, nullptr, 0, 0,
        mh, (long)L_ * D4_, DF_,
        L_, L_, D4_, L_, DF_, L_, 0, log_tau);

    // meaning = mh @ Whead^T + bhead  [BL,1000]
    wmma_gemm_tn<<<dim3(16, 8, 1), blk, 0, stream>>>(
        mh, 0, Whead, 0, 1, bhead, 0, nullptr, 0, 0, meaning, 0, 0,
        D4_, D4_, DF_, BL_, DF_, D4_, 0, log_tau);

    if (p < 2) {
      concat_x<<<BL_, blk, 0, stream>>>(feats, meaning, xbuf);
      // mu += tanh(x @ Wmu^T + bmu)
      wmma_gemm_tn<<<dim3(2, 8, 1), blk, 0, stream>>>(
          xbuf, 0, Wmu + (long)p * DS_ * D2_, 0, 1, bmu + p * DS_, 0,
          mu, 0, DS_, mu, 0, 0,
          D2_, D2_, DS_, BL_, DS_, D2_, 1, log_tau);
      gate_alpha<<<BL_, blk, 0, stream>>>(xbuf, Wg + (long)p * D2_, bg + p,
                                          alpha);
      ln_rows<<<BL_, blk, 0, stream>>>(xbuf, ln_g + (long)p * D2_,
                                       ln_b + (long)p * D2_, hbuf, D2_);
      // ffn1 = gelu(h @ W1^T + b1)   [BL,4000]  (reuses mh buffer)
      wmma_gemm_tn<<<dim3(63, 8, 1), blk, 0, stream>>>(
          hbuf, 0, W1 + (long)p * D4_ * D2_, 0, 1, b1 + (long)p * D4_, 0,
          nullptr, 0, 0, mh, 0, 0,
          D2_, D2_, D4_, BL_, D4_, D2_, 2, log_tau);
      // feats += ffn1 @ W2^T + b2
      wmma_gemm_tn<<<dim3(16, 8, 1), blk, 0, stream>>>(
          mh, 0, W2 + (long)p * DF_ * D4_, 0, 1, b2 + (long)p * DF_, 0,
          feats, 0, DF_, feats, 0, 0,
          D4_, D4_, DF_, BL_, DF_, D4_, 3, log_tau);
    }
  }

  ln_rows<<<BL_, blk, 0, stream>>>(meaning, lnf_g, lnf_b, yln, DF_);
  // out = yln @ Wlm^T   [BL,32000]
  wmma_gemm_tn<<<dim3(500, 8, 1), blk, 0, stream>>>(
      yln, 0, Wlm, 0, 1, nullptr, 0, nullptr, 0, 0, out, 0, 0,
      DF_, DF_, VV_, BL_, VV_, DF_, 0, log_tau);
}
